// Attention_6794638262338
// MI455X (gfx1250) — compile-verified, matrix operand order checked by reading, not run on hardware
//
#include <hip/hip_runtime.h>
#include <hip/hip_bf16.h>

// ---------------------------------------------------------------------------
// BitNet-style attention block on gfx1250 (MI455X), wave32 + WMMA f16->f32.
// GEMMs: 32x64 per-wave tiles, software-pipelined fragment loads.
// ---------------------------------------------------------------------------

typedef __attribute__((ext_vector_type(16))) _Float16 v16h;
typedef __attribute__((ext_vector_type(8)))  _Float16 v8h;
typedef __attribute__((ext_vector_type(8)))  float    v8f;

#define DIMV     512
#define HEADS    8
#define DHEAD    64
#define SEQ      1024
#define BATCH    8
#define ROWS     (BATCH * SEQ)      // 8192
#define INNER    (HEADS * DHEAD)    // 512
#define QKV_COLS (3 * INNER)        // 1536

// 16x32 f16 fragment loader (A layout; B uses the symmetric layout on the
// transposed operand). `p` points at (rowBlockStart, kStart) of a row-major
// matrix with leading dimension `ld` (elements). Per CDNA5 ISA 7.12.2:
//   lanes 0-15 : row = lane,    halves 0-7 => K 0-7,  halves 8-15 => K 16-23
//   lanes 16-31: row = lane-16, halves 0-7 => K 8-15, halves 8-15 => K 24-31
__device__ __forceinline__ v16h load_frag(const _Float16* p, int ld) {
  const int lane = threadIdx.x & 31;
  const int r    = lane & 15;
  const int half = lane >> 4;
  const _Float16* row = p + r * ld + half * 8;
  v8h lo = *(const v8h*)(row);
  v8h hi = *(const v8h*)(row + 16);
  v16h f;
#pragma unroll
  for (int i = 0; i < 8; ++i) { f[i] = lo[i]; f[8 + i] = hi[i]; }
  return f;
}

__device__ __forceinline__ v8f wmma_f16(v16h a, v16h b, v8f c) {
  return __builtin_amdgcn_wmma_f32_16x16x32_f16(false, a, false, b,
                                                (short)0, c, false, false);
}

// ---------------------------------------------------------------------------
// mean(|W|) reduction (single workgroup, grid-stride)
// ---------------------------------------------------------------------------
__global__ __launch_bounds__(256) void absmean_kernel(
    const float* __restrict__ W, int n, float* __restrict__ out) {
  __shared__ float red[256];
  float s = 0.f;
  for (int i = threadIdx.x; i < n; i += 256) s += fabsf(W[i]);
  red[threadIdx.x] = s;
  __syncthreads();
  for (int off = 128; off > 0; off >>= 1) {
    if (threadIdx.x < (unsigned)off) red[threadIdx.x] += red[threadIdx.x + off];
    __syncthreads();
  }
  if (threadIdx.x == 0) *out = red[0] / (float)n;
}

// ---------------------------------------------------------------------------
// ternary quantize: round(clip(w/(s+eps),-1,1)) * s  -> f16
// ---------------------------------------------------------------------------
__global__ __launch_bounds__(256) void quant_kernel(
    const float* __restrict__ W, const float* __restrict__ sptr,
    _Float16* __restrict__ Wq, int n) {
  int i = blockIdx.x * 256 + threadIdx.x;
  if (i < n) {
    float s = *sptr;
    float w = W[i] / (s + 1e-6f);
    w = fminf(1.f, fmaxf(-1.f, w));
    Wq[i] = (_Float16)(rintf(w) * s);
  }
}

// ---------------------------------------------------------------------------
// LayerNorm over 512 columns, one block per row, f32 in -> f16 out
// ---------------------------------------------------------------------------
__global__ __launch_bounds__(256) void layernorm_kernel(
    const float* __restrict__ X, const float* __restrict__ g,
    const float* __restrict__ bt, _Float16* __restrict__ Y) {
  __shared__ float red[256];
  const int row = blockIdx.x;
  const int t   = threadIdx.x;
  const float* x = X + (size_t)row * DIMV;
  float v0 = x[t], v1 = x[t + 256];

  red[t] = v0 + v1;
  __syncthreads();
  for (int off = 128; off > 0; off >>= 1) {
    if (t < off) red[t] += red[t + off];
    __syncthreads();
  }
  float mu = red[0] * (1.f / DIMV);
  __syncthreads();

  float d0 = v0 - mu, d1 = v1 - mu;
  red[t] = d0 * d0 + d1 * d1;
  __syncthreads();
  for (int off = 128; off > 0; off >>= 1) {
    if (t < off) red[t] += red[t + off];
    __syncthreads();
  }
  float rinv = rsqrtf(red[0] * (1.f / DIMV) + 1e-5f);

  _Float16* y = Y + (size_t)row * DIMV;
  y[t]       = (_Float16)(d0 * rinv * g[t] + bt[t]);
  y[t + 256] = (_Float16)(d1 * rinv * g[t + 256] + bt[t + 256]);
}

// ---------------------------------------------------------------------------
// Pipelined 32x64 wave-tile GEMM body: C[2][4] += A[32xK] * W^T tile.
// Fragments for step k+1 are issued before the 8 WMMAs of step k, so the
// compiler's s_wait_loadcnt before each WMMA batch only covers older loads.
// ---------------------------------------------------------------------------
#define GEMM_BODY(ACC, AROW0, AROW1, WBASE)                                  \
  v16h a0 = load_frag((AROW0), DIMV);                                        \
  v16h a1 = load_frag((AROW1), DIMV);                                        \
  v16h b0 = load_frag((WBASE) + (size_t)0 * 16 * DIMV, DIMV);                \
  v16h b1 = load_frag((WBASE) + (size_t)1 * 16 * DIMV, DIMV);                \
  v16h b2 = load_frag((WBASE) + (size_t)2 * 16 * DIMV, DIMV);                \
  v16h b3 = load_frag((WBASE) + (size_t)3 * 16 * DIMV, DIMV);                \
  for (int k = 32; k < DIMV; k += 32) {                                      \
    v16h na0 = load_frag((AROW0) + k, DIMV);                                 \
    v16h na1 = load_frag((AROW1) + k, DIMV);                                 \
    v16h nb0 = load_frag((WBASE) + (size_t)0 * 16 * DIMV + k, DIMV);         \
    v16h nb1 = load_frag((WBASE) + (size_t)1 * 16 * DIMV + k, DIMV);         \
    v16h nb2 = load_frag((WBASE) + (size_t)2 * 16 * DIMV + k, DIMV);         \
    v16h nb3 = load_frag((WBASE) + (size_t)3 * 16 * DIMV + k, DIMV);         \
    ACC[0][0] = wmma_f16(a0, b0, ACC[0][0]);                                 \
    ACC[1][0] = wmma_f16(a1, b0, ACC[1][0]);                                 \
    ACC[0][1] = wmma_f16(a0, b1, ACC[0][1]);                                 \
    ACC[1][1] = wmma_f16(a1, b1, ACC[1][1]);                                 \
    ACC[0][2] = wmma_f16(a0, b2, ACC[0][2]);                                 \
    ACC[1][2] = wmma_f16(a1, b2, ACC[1][2]);                                 \
    ACC[0][3] = wmma_f16(a0, b3, ACC[0][3]);                                 \
    ACC[1][3] = wmma_f16(a1, b3, ACC[1][3]);                                 \
    a0 = na0; a1 = na1; b0 = nb0; b1 = nb1; b2 = nb2; b3 = nb3;              \
  }                                                                          \
  ACC[0][0] = wmma_f16(a0, b0, ACC[0][0]);                                   \
  ACC[1][0] = wmma_f16(a1, b0, ACC[1][0]);                                   \
  ACC[0][1] = wmma_f16(a0, b1, ACC[0][1]);                                   \
  ACC[1][1] = wmma_f16(a1, b1, ACC[1][1]);                                   \
  ACC[0][2] = wmma_f16(a0, b2, ACC[0][2]);                                   \
  ACC[1][2] = wmma_f16(a1, b2, ACC[1][2]);                                   \
  ACC[0][3] = wmma_f16(a0, b3, ACC[0][3]);                                   \
  ACC[1][3] = wmma_f16(a1, b3, ACC[1][3]);

// ---------------------------------------------------------------------------
// GEMM1: qkv = xln[8192,512] @ Wqkv^T[512,1536]; scatter to Q,K (b,h,n,d)
// and V transposed (b,h,d,n), all f16. Q pre-scaled by d^-0.5.
// grid (ROWS/32, QKV_COLS/256), block 128 (4 waves, wave -> 32x64 tile)
// ---------------------------------------------------------------------------
__global__ __launch_bounds__(128) void gemm_qkv_kernel(
    const _Float16* __restrict__ X, const _Float16* __restrict__ W,
    _Float16* __restrict__ Q, _Float16* __restrict__ K,
    _Float16* __restrict__ Vt) {
  const int wave = threadIdx.x >> 5;
  const int lane = threadIdx.x & 31;
  const int r0 = blockIdx.x * 32;
  const int c0 = blockIdx.y * 256 + wave * 64;

  const _Float16* Ar0 = X + (size_t)r0 * DIMV;
  const _Float16* Ar1 = X + (size_t)(r0 + 16) * DIMV;
  const _Float16* Wb  = W + (size_t)c0 * DIMV;

  v8f acc[2][4] = {};
  GEMM_BODY(acc, Ar0, Ar1, Wb)

  const int half = lane >> 4, cl = lane & 15;
  const float qscale = 0.125f;  // DHEAD^-0.5
#pragma unroll
  for (int n = 0; n < 4; ++n) {
    const int col    = c0 + 16 * n + cl;
    const int part   = col >> 9;   // 0=q 1=k 2=v (uniform per block)
    const int within = col & 511;
    const int h = within >> 6, d = within & 63;
#pragma unroll
    for (int mt = 0; mt < 2; ++mt) {
#pragma unroll
      for (int i = 0; i < 8; ++i) {
        const int row = r0 + 16 * mt + i + 8 * half;
        const int b = row >> 10, tok = row & (SEQ - 1);
        const int bh = b * HEADS + h;
        const float v = acc[mt][n][i];
        if (part == 0)
          Q[((size_t)bh * SEQ + tok) * DHEAD + d] = (_Float16)(v * qscale);
        else if (part == 1)
          K[((size_t)bh * SEQ + tok) * DHEAD + d] = (_Float16)v;
        else
          Vt[((size_t)bh * DHEAD + d) * SEQ + tok] = (_Float16)v;
      }
    }
  }
}

// ---------------------------------------------------------------------------
// Flash attention: one block = 8 waves = 128 query rows of one (b,h).
// Wave: 16 query rows; loop key blocks of 32; scores + online softmax + PV,
// all via v_wmma_f32_16x16x32_f16. Output f32 in [b, n, h*64+d] layout.
// ---------------------------------------------------------------------------
__global__ __launch_bounds__(256) void attn_kernel(
    const _Float16* __restrict__ Q, const _Float16* __restrict__ K,
    const _Float16* __restrict__ Vt, float* __restrict__ O) {
  __shared__ _Float16 lds[8 * 16 * 32];  // per-wave 16x32 P transpose tile

  const int wave = threadIdx.x >> 5;
  const int lane = threadIdx.x & 31;
  const int bh     = blockIdx.x >> 3;   // 64 (b,h) pairs
  const int rowBlk = blockIdx.x & 7;    // 8 row-blocks of 128
  const int b = bh >> 3, h = bh & 7;
  const int r0 = rowBlk * 128 + wave * 16;

  const _Float16* Qh = Q + ((size_t)bh * SEQ + r0) * DHEAD;
  const _Float16* Kh = K + (size_t)bh * SEQ * DHEAD;
  const _Float16* Vh = Vt + (size_t)bh * DHEAD * SEQ;
  _Float16* myLds = lds + wave * (16 * 32);

  const int r = lane & 15, half = lane >> 4;

  const v16h qa0 = load_frag(Qh + 0, DHEAD);   // d = 0..31
  const v16h qa1 = load_frag(Qh + 32, DHEAD);  // d = 32..63

  v8f acc[4] = {};
  float m[8], l[8];
#pragma unroll
  for (int i = 0; i < 8; ++i) { m[i] = -1e30f; l[i] = 0.f; }

  for (int j = 0; j < SEQ; j += 32) {
    // scores for key tiles [j, j+16) and [j+16, j+32)
    v16h kb00 = load_frag(Kh + (size_t)(j + 0) * DHEAD + 0, DHEAD);
    v16h kb01 = load_frag(Kh + (size_t)(j + 0) * DHEAD + 32, DHEAD);
    v16h kb10 = load_frag(Kh + (size_t)(j + 16) * DHEAD + 0, DHEAD);
    v16h kb11 = load_frag(Kh + (size_t)(j + 16) * DHEAD + 32, DHEAD);
    v8f s0 = {}, s1 = {};
    s0 = wmma_f16(qa0, kb00, s0);
    s0 = wmma_f16(qa1, kb01, s0);
    s1 = wmma_f16(qa0, kb10, s1);
    s1 = wmma_f16(qa1, kb11, s1);

    // online softmax; row i+8*half owned by 16-lane half -> xor 1,2,4,8
    float corr[8];
#pragma unroll
    for (int i = 0; i < 8; ++i) {
      float t = fmaxf(s0[i], s1[i]);
      t = fmaxf(t, __shfl_xor(t, 1, 32));
      t = fmaxf(t, __shfl_xor(t, 2, 32));
      t = fmaxf(t, __shfl_xor(t, 4, 32));
      t = fmaxf(t, __shfl_xor(t, 8, 32));
      const float mn = fmaxf(m[i], t);
      corr[i] = __expf(m[i] - mn);
      const float p0 = __expf(s0[i] - mn);
      const float p1 = __expf(s1[i] - mn);
      float rs = p0 + p1;
      rs += __shfl_xor(rs, 1, 32);
      rs += __shfl_xor(rs, 2, 32);
      rs += __shfl_xor(rs, 4, 32);
      rs += __shfl_xor(rs, 8, 32);
      l[i] = l[i] * corr[i] + rs;
      m[i] = mn;
      s0[i] = p0;
      s1[i] = p1;
    }
#pragma unroll
    for (int n = 0; n < 4; ++n)
#pragma unroll
      for (int i = 0; i < 8; ++i) acc[n][i] *= corr[i];

    // C-frag (col-owned) -> A-frag (row-owned) transpose via wave-private LDS
#pragma unroll
    for (int i = 0; i < 8; ++i) {
      myLds[(i + 8 * half) * 32 + r]      = (_Float16)s0[i];
      myLds[(i + 8 * half) * 32 + 16 + r] = (_Float16)s1[i];
    }
    asm volatile("s_wait_dscnt 0" ::: "memory");  // wave-local LDS RAW fence
    const v16h pa = load_frag(myLds, 32);

    // acc += P(16x32) @ V(32x64); Vt rows are d, contiguous in key index
#pragma unroll
    for (int n = 0; n < 4; ++n) {
      v16h vb = load_frag(Vh + (size_t)(16 * n) * SEQ + j, SEQ);
      acc[n] = wmma_f16(pa, vb, acc[n]);
    }
  }

  // normalize and store: out[b, tok, h*64 + d]
  float inv[8];
#pragma unroll
  for (int i = 0; i < 8; ++i) inv[i] = 1.f / l[i];
#pragma unroll
  for (int n = 0; n < 4; ++n) {
    const int d = 16 * n + r;
#pragma unroll
    for (int i = 0; i < 8; ++i) {
      const int tok = r0 + i + 8 * half;
      O[((size_t)b * SEQ + tok) * INNER + h * DHEAD + d] = acc[n][i] * inv[i];
    }
  }
}

// ---------------------------------------------------------------------------
// GEMM2: out = ln2[8192,512] @ Wout^T[512,512] + b_out -> f32
// grid (ROWS/32, 512/256), block 128, wave -> 32x64 tile
// ---------------------------------------------------------------------------
__global__ __launch_bounds__(128) void gemm_out_kernel(
    const _Float16* __restrict__ X, const _Float16* __restrict__ W,
    const float* __restrict__ bias, float* __restrict__ O) {
  const int wave = threadIdx.x >> 5;
  const int lane = threadIdx.x & 31;
  const int r0 = blockIdx.x * 32;
  const int c0 = blockIdx.y * 256 + wave * 64;

  const _Float16* Ar0 = X + (size_t)r0 * DIMV;
  const _Float16* Ar1 = X + (size_t)(r0 + 16) * DIMV;
  const _Float16* Wb  = W + (size_t)c0 * DIMV;

  v8f acc[2][4] = {};
  GEMM_BODY(acc, Ar0, Ar1, Wb)

  const int half = lane >> 4, cl = lane & 15;
#pragma unroll
  for (int n = 0; n < 4; ++n) {
    const int col = c0 + 16 * n + cl;
    const float bb = bias[col];
#pragma unroll
    for (int mt = 0; mt < 2; ++mt) {
#pragma unroll
      for (int i = 0; i < 8; ++i) {
        const int row = r0 + 16 * mt + i + 8 * half;
        O[(size_t)row * DIMV + col] = acc[mt][n][i] + bb;
      }
    }
  }
}

// ---------------------------------------------------------------------------
extern "C" void kernel_launch(void* const* d_in, const int* in_sizes, int n_in,
                              void* d_out, int out_size, void* d_ws,
                              size_t ws_size, hipStream_t stream) {
  const float* x     = (const float*)d_in[0];
  const float* g1    = (const float*)d_in[1];
  const float* b1    = (const float*)d_in[2];
  const float* W_qkv = (const float*)d_in[3];
  const float* g2    = (const float*)d_in[4];
  const float* b2    = (const float*)d_in[5];
  const float* W_out = (const float*)d_in[6];
  const float* b_out = (const float*)d_in[7];
  float* out = (float*)d_out;

  char* ws = (char*)d_ws;
  size_t off = 0;
  auto alloc = [&](size_t bytes) -> char* {
    char* p = ws + off;
    off = (off + bytes + 255) & ~(size_t)255;
    return p;
  };

  float*     scales = (float*)alloc(8);
  _Float16*  wqkv_q = (_Float16*)alloc((size_t)QKV_COLS * DIMV * 2);
  _Float16*  wout_q = (_Float16*)alloc((size_t)DIMV * INNER * 2);
  _Float16*  xln    = (_Float16*)alloc((size_t)ROWS * DIMV * 2);
  _Float16*  Qb     = (_Float16*)alloc((size_t)BATCH * HEADS * SEQ * DHEAD * 2);
  _Float16*  Kb     = (_Float16*)alloc((size_t)BATCH * HEADS * SEQ * DHEAD * 2);
  _Float16*  Vtb    = (_Float16*)alloc((size_t)BATCH * HEADS * SEQ * DHEAD * 2);
  float*     attO   = (float*)alloc((size_t)ROWS * INNER * 4);
  _Float16*  ln2    = (_Float16*)alloc((size_t)ROWS * INNER * 2);

  absmean_kernel<<<1, 256, 0, stream>>>(W_qkv, QKV_COLS * DIMV, scales + 0);
  absmean_kernel<<<1, 256, 0, stream>>>(W_out, DIMV * INNER, scales + 1);

  quant_kernel<<<(QKV_COLS * DIMV + 255) / 256, 256, 0, stream>>>(
      W_qkv, scales + 0, wqkv_q, QKV_COLS * DIMV);
  quant_kernel<<<(DIMV * INNER + 255) / 256, 256, 0, stream>>>(
      W_out, scales + 1, wout_q, DIMV * INNER);

  layernorm_kernel<<<ROWS, 256, 0, stream>>>(x, g1, b1, xln);

  gemm_qkv_kernel<<<dim3(ROWS / 32, QKV_COLS / 256), 128, 0, stream>>>(
      xln, wqkv_q, Qb, Kb, Vtb);

  attn_kernel<<<BATCH * HEADS * (SEQ / 128), 256, 0, stream>>>(Qb, Kb, Vtb,
                                                               attO);

  layernorm_kernel<<<ROWS, 256, 0, stream>>>(attO, g2, b2, ln2);

  gemm_out_kernel<<<dim3(ROWS / 32, INNER / 256), 128, 0, stream>>>(
      ln2, wout_q, b_out, out);
}
